// MixtureOfExperts_7387343749154
// MI455X (gfx1250) — compile-verified, matrix operand order checked by reading, not run on hardware
//
#include <hip/hip_runtime.h>
#include <hip/hip_bf16.h>
#include <math.h>

typedef _Float16 f16;
typedef __attribute__((ext_vector_type(16))) _Float16 v16h;
typedef __attribute__((ext_vector_type(8)))  _Float16 v8h;
typedef __attribute__((ext_vector_type(8)))  float    v8f;
typedef __attribute__((ext_vector_type(4)))  float    v4f;
typedef __attribute__((ext_vector_type(4)))  unsigned int v4u;
typedef __attribute__((ext_vector_type(8)))  int      v8i;
typedef __attribute__((ext_vector_type(4)))  int      v4i;

#define MOE_D 1024
#define MOE_H 4096
#define MOE_E 8
#define LDA   40          // padded LDS row stride in halves (80B)
#define TILE_BYTES 10240  // 128 * LDA * 2

__device__ __forceinline__ v16h cat8(v8h lo, v8h hi) {
  return __builtin_shufflevector(lo, hi, 0,1,2,3,4,5,6,7,8,9,10,11,12,13,14,15);
}

__device__ __forceinline__ float gelu_tanh(float v) {
  const float c = 0.7978845608028654f;  // sqrt(2/pi)
  return 0.5f * v * (1.0f + tanhf(c * (v + 0.044715f * v * v * v)));
}

// ---- Tensor Data Mover: 2D tile (128 rows x 32 f16) -> padded LDS rows (80B) ----
// D# per CDNA5 ISA ch.8: group0 = {flags, lds_addr, global_addr, type=2},
// group1 = {data_size=2B, pad 64B->+16B (interval=16 DW, amount=4 DW), dims, strides}.
// 6-arg builtin form (clang-23 amdgpu-toolchain): (v4u, v8i, v4i, v4i, v8i, i32 cpol).
__device__ __forceinline__ void tdm_load_2d(unsigned lds_off, unsigned long long gaddr,
                                            unsigned tdim0, unsigned tdim1,
                                            unsigned long long stride0) {
  v4u g0;
  g0[0] = 1u;                                   // count=1, user descriptor
  g0[1] = lds_off;                              // LDS byte address
  g0[2] = (unsigned)gaddr;                      // global_addr[31:0]
  g0[3] = (unsigned)(gaddr >> 32) | (2u << 30); // global_addr[56:32] | type=2
  v8i g1;
  g1[0] = (int)((1u << 16) | (1u << 20) | (3u << 22) | (3u << 25)); // 2B elems, pad on
  g1[1] = (int)((tdim0 & 0xFFFFu) << 16);                           // tensor_dim0[15:0]
  g1[2] = (int)((tdim0 >> 16) | ((tdim1 & 0xFFFFu) << 16));         // dim0 hi | dim1 lo
  g1[3] = (int)((tdim1 >> 16) | (32u << 16));                       // dim1 hi | tile_dim0=32
  g1[4] = (int)128;                                                 // tile_dim1=128
  g1[5] = (int)(unsigned)stride0;                                   // dim0 stride lo32
  g1[6] = (int)((unsigned)(stride0 >> 32) & 0xFFFFu);               // stride hi16
  g1[7] = 0;
  v4i z4 = {0, 0, 0, 0};                         // 2D tensor: groups 2/3 unused
  v8i z8 = {0, 0, 0, 0, 0, 0, 0, 0};             // trailing group (clang-23 form)
  __builtin_amdgcn_tensor_load_to_lds(g0, g1, z4, z4, z8, 0);
}

// ---------------- zero-init: out and per-expert counters ----------------
__global__ void moe_zero(float* __restrict__ out, size_t n, int* __restrict__ counts) {
  size_t i = (size_t)blockIdx.x * blockDim.x + threadIdx.x;
  if (i < MOE_E) counts[i] = 0;
  for (size_t k = i; k < n; k += (size_t)gridDim.x * blockDim.x) out[k] = 0.0f;
}

// ---------------- weight convert+transpose: fp32 [K,N] -> f16 [N,K] ----------------
__global__ void __launch_bounds__(256)
moe_transpose_cvt(const float* __restrict__ W, f16* __restrict__ Wt, int K, int Nn) {
  __shared__ float S[32][33];
  const int e = blockIdx.z;
  const float* __restrict__ We = W + (size_t)e * K * Nn;
  f16* __restrict__ Wte = Wt + (size_t)e * K * Nn;
  const int n0 = blockIdx.x * 32, k0 = blockIdx.y * 32;
  const int tx = threadIdx.x & 31, ty = threadIdx.x >> 5;  // ty 0..7
  #pragma unroll
  for (int i = 0; i < 4; ++i)
    S[ty + 8 * i][tx] = We[(size_t)(k0 + ty + 8 * i) * Nn + n0 + tx];  // coalesced in n
  __syncthreads();
  #pragma unroll
  for (int i = 0; i < 4; ++i)
    Wte[(size_t)(n0 + ty + 8 * i) * K + k0 + tx] = (f16)S[tx][ty + 8 * i];  // coalesced in k
}

// ---------------- router: logits -> softmax -> top-2 gates ----------------
__global__ void __launch_bounds__(256)
moe_router(const float* __restrict__ x, const float* __restrict__ Wr,
           float* __restrict__ gates, int N) {
  __shared__ float WrS[MOE_D * MOE_E];   // 32 KB
  const int t = threadIdx.x;
  for (int i = t; i < MOE_D * MOE_E; i += 256) WrS[i] = Wr[i];
  __syncthreads();
  const int tok = blockIdx.x * 256 + t;
  if (tok >= N) return;
  const float* __restrict__ xr = x + (size_t)tok * MOE_D;
  float acc[MOE_E] = {};
  for (int d = 0; d < MOE_D; ++d) {
    float xv = xr[d];
    #pragma unroll
    for (int e = 0; e < MOE_E; ++e) acc[e] = fmaf(xv, WrS[d * MOE_E + e], acc[e]);
  }
  float m = acc[0];
  #pragma unroll
  for (int e = 1; e < MOE_E; ++e) m = fmaxf(m, acc[e]);
  float p[MOE_E]; float s = 0.0f;
  #pragma unroll
  for (int e = 0; e < MOE_E; ++e) { p[e] = __expf(acc[e] - m); s += p[e]; }
  const float inv = 1.0f / s;
  int i1 = 0;
  #pragma unroll
  for (int e = 1; e < MOE_E; ++e) if (p[e] > p[i1]) i1 = e;   // first index wins ties
  int i2 = (i1 == 0) ? 1 : 0;
  #pragma unroll
  for (int e = 0; e < MOE_E; ++e) if (e != i1 && e != i2 && p[e] > p[i2]) i2 = e;
  #pragma unroll
  for (int e = 0; e < MOE_E; ++e)
    gates[(size_t)tok * MOE_E + e] = (e == i1 || e == i2) ? p[e] * inv : 0.0f;
}

// ---------------- build per-expert compacted token lists ----------------
__global__ void __launch_bounds__(256)
moe_assign(const float* __restrict__ gates, int* __restrict__ counts,
           int* __restrict__ lists, int N) {
  const int tok = blockIdx.x * 256 + threadIdx.x;
  if (tok >= N) return;
  #pragma unroll
  for (int e = 0; e < MOE_E; ++e) {
    float g = gates[(size_t)tok * MOE_E + e];
    if (g > 0.0f) {
      int slot = atomicAdd(&counts[e], 1);
      lists[(size_t)e * N + slot] = tok;
    }
  }
}

// ---------------- GEMM1: h = gelu(gather(x) @ W1[e] + b1[e]), f16 out ----------------
// Dynamic LDS layout: As @ 0, Bs[0] @ 10240, Bs[1] @ 20480, tokS @ 30720  (31232 B)
__global__ void __launch_bounds__(256)
moe_gemm1(const float* __restrict__ x, const f16* __restrict__ W1t,
          const float* __restrict__ b1, const int* __restrict__ lists,
          const int* __restrict__ counts, f16* __restrict__ h, int N, int e) {
  extern __shared__ __align__(16) char smem[];
  f16* As  = (f16*)smem;
  int* tokS = (int*)(smem + 3 * TILE_BYTES);

  const int cnt = counts[e];
  const int m0 = blockIdx.y * 128;
  if (m0 >= cnt) return;                 // gathered top-2: skip inactive tiles
  const int n0 = blockIdx.x * 128;
  const f16* __restrict__ W1te = W1t + (size_t)e * MOE_H * MOE_D;  // [H][D] f16

  const int t = threadIdx.x;
  if (t < 128) {
    int r = m0 + t;
    tokS[t] = (r < cnt) ? lists[(size_t)e * N + r] : lists[(size_t)e * N];
  }
  __syncthreads();

  const int lane = t & 31;
  const int wr   = (t >> 5) >> 2;        // wave row 0..1
  const int wc   = (t >> 5) & 3;         // wave col 0..3
  const int lm   = lane & 15;
  const int kb   = (lane >> 4) * 8;      // frag K base (ISA 16-bit A layout)
  const int rsel = (lane >> 4) * 8;      // C/D: M = v + 8*(lane>=16)

  v8f acc[4][2];
  #pragma unroll
  for (int i = 0; i < 4; ++i)
    #pragma unroll
    for (int j = 0; j < 2; ++j)
      acc[i][j] = v8f{0.f,0.f,0.f,0.f,0.f,0.f,0.f,0.f};

  const int ar = t >> 1;                 // A stage: row
  const int ah = (t & 1) * 16;           // A stage: k offset 0/16
  const float* __restrict__ xrow = x + (size_t)tokS[ar] * MOE_D;

  // TDM double-buffered weight tile: Wt rows n0..n0+127, k slice, row stride D
  const unsigned long long bBase =
      (unsigned long long)(uintptr_t)W1te + (size_t)n0 * MOE_D * 2;
  if (t < 32) tdm_load_2d(TILE_BYTES, bBase, MOE_D, MOE_H, MOE_D);

  int buf = 0;
  for (int kk = 0; kk < MOE_D; kk += 32) {
    const bool more = (kk + 32) < MOE_D;
    if (t < 32 && more)
      tdm_load_2d(TILE_BYTES + (unsigned)((buf ^ 1) * TILE_BYTES),
                  bBase + (size_t)(kk + 32) * 2, MOE_D, MOE_H, MOE_D);
    { // stage A: gathered fp32 x rows -> f16 LDS
      const float* p = xrow + kk + ah;
      v4f a0 = *(const v4f*)(p);      v4f a1 = *(const v4f*)(p + 4);
      v4f a2 = *(const v4f*)(p + 8);  v4f a3 = *(const v4f*)(p + 12);
      v8h lo = {(f16)a0.x,(f16)a0.y,(f16)a0.z,(f16)a0.w,(f16)a1.x,(f16)a1.y,(f16)a1.z,(f16)a1.w};
      v8h hi = {(f16)a2.x,(f16)a2.y,(f16)a2.z,(f16)a2.w,(f16)a3.x,(f16)a3.y,(f16)a3.z,(f16)a3.w};
      *(v8h*)&As[ar * LDA + ah]     = lo;
      *(v8h*)&As[ar * LDA + ah + 8] = hi;
      if (more) __builtin_prefetch(p + 32, 0, 1);
    }
    if (t < 32) {
      if (more) __builtin_amdgcn_s_wait_tensorcnt(1);   // current tile done
      else      __builtin_amdgcn_s_wait_tensorcnt(0);
    }
    __syncthreads();

    const f16* Bs = (const f16*)(smem + TILE_BYTES + buf * TILE_BYTES);
    v16h af[4];
    #pragma unroll
    for (int i = 0; i < 4; ++i) {
      int m = wr * 64 + i * 16 + lm;
      af[i] = cat8(*(const v8h*)&As[m * LDA + kb], *(const v8h*)&As[m * LDA + kb + 16]);
    }
    v16h bf[2];
    #pragma unroll
    for (int j = 0; j < 2; ++j) {
      int n = wc * 32 + j * 16 + lm;
      bf[j] = cat8(*(const v8h*)&Bs[n * LDA + kb], *(const v8h*)&Bs[n * LDA + kb + 16]);
    }
    #pragma unroll
    for (int i = 0; i < 4; ++i)
      #pragma unroll
      for (int j = 0; j < 2; ++j)
        acc[i][j] = __builtin_amdgcn_wmma_f32_16x16x32_f16(
            false, af[i], false, bf[j], (short)0, acc[i][j], false, false);
    __syncthreads();
    buf ^= 1;
  }

  // epilogue: + bias, gelu, store f16 h at compacted row index
  #pragma unroll
  for (int j = 0; j < 2; ++j) {
    const int n = n0 + wc * 32 + j * 16 + lm;
    const float bias = b1[(size_t)e * MOE_H + n];
    #pragma unroll
    for (int i = 0; i < 4; ++i)
      #pragma unroll
      for (int v = 0; v < 8; ++v) {
        int m = m0 + wr * 64 + i * 16 + rsel + v;
        h[(size_t)m * MOE_H + n] = (f16)gelu_tanh(acc[i][j][v] + bias);
      }
  }
}

// ---------------- GEMM2: out[tok] += gate * (h @ W2[e] + b2[e]) ----------------
// Dynamic LDS: As[0]@0, As[1]@10240, Bs[0]@20480, Bs[1]@30720, tokS@40960, gateS@41472
__global__ void __launch_bounds__(256)
moe_gemm2(const f16* __restrict__ h, const f16* __restrict__ W2t,
          const float* __restrict__ b2, const float* __restrict__ gates,
          const int* __restrict__ lists, const int* __restrict__ counts,
          float* __restrict__ out, int N, int e) {
  extern __shared__ __align__(16) char smem[];
  int*   tokS  = (int*)(smem + 4 * TILE_BYTES);
  float* gateS = (float*)(smem + 4 * TILE_BYTES + 512);

  const int cnt = counts[e];
  const int m0 = blockIdx.y * 128;
  if (m0 >= cnt) return;
  const int n0 = blockIdx.x * 128;
  const f16* __restrict__ W2te = W2t + (size_t)e * MOE_D * MOE_H;  // [D][H] f16

  const int t = threadIdx.x;
  if (t < 128) {
    int r = m0 + t;
    if (r < cnt) {
      int tk = lists[(size_t)e * N + r];
      tokS[t]  = tk;
      gateS[t] = gates[(size_t)tk * MOE_E + e];
    } else { tokS[t] = -1; gateS[t] = 0.0f; }
  }

  const int lane = t & 31;
  const int wr   = (t >> 5) >> 2;
  const int wc   = (t >> 5) & 3;
  const int lm   = lane & 15;
  const int kb   = (lane >> 4) * 8;
  const int rsel = (lane >> 4) * 8;

  v8f acc[4][2];
  #pragma unroll
  for (int i = 0; i < 4; ++i)
    #pragma unroll
    for (int j = 0; j < 2; ++j)
      acc[i][j] = v8f{0.f,0.f,0.f,0.f,0.f,0.f,0.f,0.f};

  // TDM double-buffered: A = compacted h rows [m0..m0+127] (stride H),
  //                      B = W2t rows [n0..n0+127] (stride H)
  const unsigned long long aBase =
      (unsigned long long)(uintptr_t)h + (size_t)m0 * MOE_H * 2;
  const unsigned long long bBase =
      (unsigned long long)(uintptr_t)W2te + (size_t)n0 * MOE_H * 2;
  if (t < 32) {
    tdm_load_2d(0, aBase, MOE_H, (unsigned)N, MOE_H);
    tdm_load_2d(2 * TILE_BYTES, bBase, MOE_H, MOE_D, MOE_H);
  }

  int buf = 0;
  for (int kk = 0; kk < MOE_H; kk += 32) {
    const bool more = (kk + 32) < MOE_H;
    if (t < 32) {
      if (more) {
        tdm_load_2d((unsigned)((buf ^ 1) * TILE_BYTES),
                    aBase + (size_t)(kk + 32) * 2, MOE_H, (unsigned)N, MOE_H);
        tdm_load_2d((unsigned)(2 * TILE_BYTES + (buf ^ 1) * TILE_BYTES),
                    bBase + (size_t)(kk + 32) * 2, MOE_H, MOE_D, MOE_H);
        __builtin_amdgcn_s_wait_tensorcnt(2);   // current A+B done, next pair in flight
      } else {
        __builtin_amdgcn_s_wait_tensorcnt(0);
      }
    }
    __syncthreads();

    const f16* As = (const f16*)(smem + buf * TILE_BYTES);
    const f16* Bs = (const f16*)(smem + 2 * TILE_BYTES + buf * TILE_BYTES);
    v16h af[4];
    #pragma unroll
    for (int i = 0; i < 4; ++i) {
      int m = wr * 64 + i * 16 + lm;
      af[i] = cat8(*(const v8h*)&As[m * LDA + kb], *(const v8h*)&As[m * LDA + kb + 16]);
    }
    v16h bf[2];
    #pragma unroll
    for (int j = 0; j < 2; ++j) {
      int n = wc * 32 + j * 16 + lm;
      bf[j] = cat8(*(const v8h*)&Bs[n * LDA + kb], *(const v8h*)&Bs[n * LDA + kb + 16]);
    }
    #pragma unroll
    for (int i = 0; i < 4; ++i)
      #pragma unroll
      for (int j = 0; j < 2; ++j)
        acc[i][j] = __builtin_amdgcn_wmma_f32_16x16x32_f16(
            false, af[i], false, bf[j], (short)0, acc[i][j], false, false);
    __syncthreads();
    buf ^= 1;
  }

  // epilogue: gate-weighted scatter accumulate (race-free: token unique per expert,
  // experts are sequential launches on the stream)
  #pragma unroll
  for (int j = 0; j < 2; ++j) {
    const int n = n0 + wc * 32 + j * 16 + lm;
    const float bias = b2[(size_t)e * MOE_D + n];
    #pragma unroll
    for (int i = 0; i < 4; ++i)
      #pragma unroll
      for (int v = 0; v < 8; ++v) {
        int ml = wr * 64 + i * 16 + rsel + v;
        int tk = tokS[ml];
        if (tk < 0) continue;
        float y = acc[i][j][v] + bias;
        float* o = out + (size_t)tk * MOE_D + n;
        *o += gateS[ml] * y;
      }
  }
}

extern "C" void kernel_launch(void* const* d_in, const int* in_sizes, int n_in,
                              void* d_out, int out_size, void* d_ws, size_t ws_size,
                              hipStream_t stream) {
  const float* x  = (const float*)d_in[0];
  const float* Wr = (const float*)d_in[1];
  const float* W1 = (const float*)d_in[2];
  const float* b1 = (const float*)d_in[3];
  const float* W2 = (const float*)d_in[4];
  const float* b2 = (const float*)d_in[5];
  float* out = (float*)d_out;

  const int N = in_sizes[0] / MOE_D;     // 8192 tokens

  // workspace layout
  char* ws = (char*)d_ws;
  float* gates = (float*)ws;                                    // N*8 f32
  size_t off = (size_t)N * MOE_E * sizeof(float);
  int* counts = (int*)(ws + off);  off += 256;                  // 8 ints (padded)
  int* lists  = (int*)(ws + off);  off += (size_t)MOE_E * N * sizeof(int);
  off = (off + 255) & ~(size_t)255;
  f16* h = (f16*)(ws + off);       off += (size_t)N * MOE_H * sizeof(f16);
  off = (off + 255) & ~(size_t)255;
  f16* W1t = (f16*)(ws + off);     off += (size_t)MOE_E * MOE_D * MOE_H * sizeof(f16);
  off = (off + 255) & ~(size_t)255;
  f16* W2t = (f16*)(ws + off);                                  // [E][D][H] f16

  const size_t outElems = (size_t)N * MOE_D;
  moe_zero<<<1024, 256, 0, stream>>>(out, outElems, counts);

  // one-shot weight convert + transpose (fp32 [K,N] -> f16 [N,K])
  dim3 tg1(MOE_H / 32, MOE_D / 32, MOE_E);   // W1 [D,H] -> W1t [H,D]
  moe_transpose_cvt<<<tg1, 256, 0, stream>>>(W1, W1t, MOE_D, MOE_H);
  dim3 tg2(MOE_D / 32, MOE_H / 32, MOE_E);   // W2 [H,D] -> W2t [D,H]
  moe_transpose_cvt<<<tg2, 256, 0, stream>>>(W2, W2t, MOE_H, MOE_D);

  moe_router<<<(N + 255) / 256, 256, 0, stream>>>(x, Wr, gates, N);
  moe_assign<<<(N + 255) / 256, 256, 0, stream>>>(gates, counts, lists, N);

  dim3 g1(MOE_H / 128, N / 128);   // (32, 64) — blocks early-exit past count[e]
  dim3 g2(MOE_D / 128, N / 128);   // (8, 64)
  const size_t lds1 = 3 * TILE_BYTES + 512;          // As + Bs*2 + tokS
  const size_t lds2 = 4 * TILE_BYTES + 1024;         // As*2 + Bs*2 + tokS + gateS
  for (int e = 0; e < MOE_E; ++e) {
    moe_gemm1<<<g1, 256, lds1, stream>>>(x, W1t, b1, lists, counts, h, N, e);
    moe_gemm2<<<g2, 256, lds2, stream>>>(h, W2t, b2, gates, lists, counts, out, N, e);
  }
}
